// WorkingLSTM_43568148251391
// MI455X (gfx1250) — compile-verified, matrix operand order checked by reading, not run on hardware
//
#include <hip/hip_runtime.h>

#define B_TOT 1024
#define T_LEN 512
#define I_DIM 6
#define IP    8     // x K-dim padded to 8 for 2x WMMA 16x16x4
#define H_DIM 64
#define G_DIM 256   // 4*H
#define BM    16    // batch rows per block

typedef __attribute__((ext_vector_type(16))) __bf16 v16bf;
typedef __attribute__((ext_vector_type(8)))  __bf16 v8bf;
typedef __attribute__((ext_vector_type(8)))  float  v8f;
typedef __attribute__((ext_vector_type(2)))  float  v2f;

__device__ __forceinline__ float fast_sigmoid(float x) {
    return __builtin_amdgcn_rcpf(1.0f + __expf(-x));
}
__device__ __forceinline__ float fast_tanh(float x) {
    float e = __expf(2.0f * x);
    return 1.0f - 2.0f * __builtin_amdgcn_rcpf(e + 1.0f);
}

__device__ __forceinline__ v8f wmma_bf16(v16bf a, v16bf b, v8f c) {
    // D = A(16x32 bf16) * B(32x16 bf16) + C(16x16 f32)
    return __builtin_amdgcn_wmma_f32_16x16x32_bf16(
        false, a, false, b, (short)0, c, false, false);
}
__device__ __forceinline__ v8f wmma_f32x4(v2f a, v2f b, v8f c) {
    // D = A(16x4 f32) * B(4x16 f32) + C(16x16 f32)
    return __builtin_amdgcn_wmma_f32_16x16x4_f32(
        false, a, false, b, (short)0, c, false, false);
}

// Load one 16x32 bf16 fragment from a row-major [rows][64] bf16 buffer.
// ISA layout (CDNA5 ISA): lane<16 holds K {kh*32+0..7, kh*32+16..23},
// lane>=16 holds K {kh*32+8..15, kh*32+24..31}. Two 16B chunks -> 2x ds_load_b128.
__device__ __forceinline__ v16bf load_frag(const __bf16* base, int row, int kh, int hi) {
    union { v16bf v; v8bf h[2]; } u;
    const __bf16* p = base + row * 64 + kh * 32 + hi * 8;
    u.h[0] = *(const v8bf*)(p);
    u.h[1] = *(const v8bf*)(p + 16);
    return u.v;
}

// Fused 2-layer LSTM over all T steps. Grid: B/16 blocks, 128 threads (4 waves).
// Wave w owns h-column tile j=w (cols 16w..16w+15) and gate tiles nt = g*4+w.
__global__ void __launch_bounds__(128, 1)
lstm2_fused_kernel(const float* __restrict__ x,
                   const float* __restrict__ Wih0,
                   const float* __restrict__ Whh0,
                   const float* __restrict__ bih0,
                   const float* __restrict__ bhh0,
                   const float* __restrict__ Wih1,
                   const float* __restrict__ Whh1,
                   const float* __restrict__ bih1,
                   const float* __restrict__ bhh1,
                   float* __restrict__ h_last)
{
    extern __shared__ char smem[];
    __bf16* sWhh0  = (__bf16*)smem;                      // 256*64 bf16 = 32KB
    __bf16* sWih1  = sWhh0 + G_DIM * H_DIM;              // 32KB
    __bf16* sWhh1  = sWih1 + G_DIM * H_DIM;              // 32KB
    float*  sB0    = (float*)(sWhh1 + G_DIM * H_DIM);    // 256 f32
    float*  sB1    = sB0 + G_DIM;                        // 256 f32
    float*  sWih0p = sB1 + G_DIM;                        // 256*8 f32 (K padded)
    float*  sX     = sWih0p + G_DIM * IP;                // 16*8 f32 (K padded)
    __bf16* sH1    = (__bf16*)(sX + BM * IP);            // 16*64 bf16 (16B aligned)
    __bf16* sH2    = sH1 + BM * H_DIM;                   // 16*64 bf16

    const int tid = threadIdx.x;

    // ---- one-time weight preload into LDS ----
    for (int i = tid; i < G_DIM * H_DIM; i += 128) {
        sWhh0[i] = (__bf16)Whh0[i];
        sWih1[i] = (__bf16)Wih1[i];
        sWhh1[i] = (__bf16)Whh1[i];
    }
    for (int i = tid; i < G_DIM; i += 128) {
        sB0[i] = bih0[i] + bhh0[i];
        sB1[i] = bih1[i] + bhh1[i];
    }
    for (int i = tid; i < G_DIM * IP; i += 128) {        // Wih0 zero-padded K=6->8
        const int r = i >> 3, c = i & 7;
        sWih0p[i] = (c < I_DIM) ? Wih0[r * I_DIM + c] : 0.0f;
    }
    __syncthreads();

    const int lane = tid & 31;
    const int wv   = tid >> 5;      // 0..3 : this wave's h-column tile
    const int hi   = lane >> 4;     // lane half
    const int lc   = lane & 15;     // row/col within 16-wide tile
    const int b0   = blockIdx.x * BM;
    const int colg = wv * 16 + lc;  // h column this lane produces

    // per-wave bias registers for its 4 gate tiles per layer
    float bias0r[4], bias1r[4];
    #pragma unroll
    for (int g = 0; g < 4; ++g) {
        const int col = (g * 4 + wv) * 16 + lc;
        bias0r[g] = sB0[col];
        bias1r[g] = sB1[col];
    }

    // recurrent state: h fragments (bf16 A-layout) + c tiles (f32 D-layout)
    v16bf h1f[2], h2f[2];
    #pragma unroll
    for (int k = 0; k < 2; ++k)
        for (int e = 0; e < 16; ++e) { h1f[k][e] = (__bf16)0.0f; h2f[k][e] = (__bf16)0.0f; }
    float c1r[8] = {}, c2r[8] = {};

    #pragma unroll 1
    for (int t = 0; t < T_LEN; ++t) {
        // ---- stage x_t tile (16 x 8, zero-padded) into LDS ----
        __syncthreads();
        {
            const int r = tid >> 3, c = tid & 7;   // 128 threads = 16*8
            sX[tid] = (c < I_DIM)
                ? x[((size_t)(b0 + r) * T_LEN + t) * I_DIM + c] : 0.0f;
        }
        __syncthreads();

        // x A-fragments for WMMA 16x16x4 (32-bit A: lane<16 K={0,1}, lane>=16 K={2,3})
        const v2f xa0 = *(const v2f*)(sX + lc * IP + 0 + hi * 2);
        const v2f xa1 = *(const v2f*)(sX + lc * IP + 4 + hi * 2);

        // ================= layer 1 =================
        v8f acc[4];
        #pragma unroll
        for (int g = 0; g < 4; ++g) {
            const int nt  = g * 4 + wv;
            const int row = nt * 16 + lc;
            v8f a;
            #pragma unroll
            for (int v = 0; v < 8; ++v) a[v] = bias0r[g];
            // x @ Wih0^T  (K=8 as 2x K=4, f32 exact)
            const v2f wb0 = *(const v2f*)(sWih0p + row * IP + 0 + hi * 2);
            const v2f wb1 = *(const v2f*)(sWih0p + row * IP + 4 + hi * 2);
            a = wmma_f32x4(xa0, wb0, a);
            a = wmma_f32x4(xa1, wb1, a);
            // h1 @ Whh0^T (K=64 as 2x K=32, bf16)
            a = wmma_bf16(h1f[0], load_frag(sWhh0, row, 0, hi), a);
            a = wmma_bf16(h1f[1], load_frag(sWhh0, row, 1, hi), a);
            acc[g] = a;
        }
        float hn1[8];
        #pragma unroll
        for (int v = 0; v < 8; ++v) {
            const float ig = fast_sigmoid(acc[0][v]);
            const float fg = fast_sigmoid(acc[1][v]);
            const float gg = fast_tanh(acc[2][v]);
            const float og = fast_sigmoid(acc[3][v]);
            const float c  = fg * c1r[v] + ig * gg;
            c1r[v] = c;
            hn1[v] = og * fast_tanh(c);
        }
        __syncthreads();
        #pragma unroll
        for (int v = 0; v < 8; ++v)
            sH1[(hi * 8 + v) * H_DIM + colg] = (__bf16)hn1[v];
        __syncthreads();
        h1f[0] = load_frag(sH1, lc, 0, hi);
        h1f[1] = load_frag(sH1, lc, 1, hi);

        // ================= layer 2 =================
        #pragma unroll
        for (int g = 0; g < 4; ++g) {
            const int nt  = g * 4 + wv;
            const int row = nt * 16 + lc;
            v8f a;
            #pragma unroll
            for (int v = 0; v < 8; ++v) a[v] = bias1r[g];
            a = wmma_bf16(h1f[0], load_frag(sWih1, row, 0, hi), a);
            a = wmma_bf16(h1f[1], load_frag(sWih1, row, 1, hi), a);
            a = wmma_bf16(h2f[0], load_frag(sWhh1, row, 0, hi), a);
            a = wmma_bf16(h2f[1], load_frag(sWhh1, row, 1, hi), a);
            acc[g] = a;
        }
        float hn2[8];
        #pragma unroll
        for (int v = 0; v < 8; ++v) {
            const float ig = fast_sigmoid(acc[0][v]);
            const float fg = fast_sigmoid(acc[1][v]);
            const float gg = fast_tanh(acc[2][v]);
            const float og = fast_sigmoid(acc[3][v]);
            const float c  = fg * c2r[v] + ig * gg;
            c2r[v] = c;
            hn2[v] = og * fast_tanh(c);
        }
        if (t == T_LEN - 1) {
            #pragma unroll
            for (int v = 0; v < 8; ++v)
                h_last[(size_t)(b0 + hi * 8 + v) * H_DIM + colg] = hn2[v];
        } else {
            __syncthreads();
            #pragma unroll
            for (int v = 0; v < 8; ++v)
                sH2[(hi * 8 + v) * H_DIM + colg] = (__bf16)hn2[v];
            __syncthreads();
            h2f[0] = load_frag(sH2, lc, 0, hi);
            h2f[1] = load_frag(sH2, lc, 1, hi);
        }
    }
}

// Tiny MLP head: one thread per batch row (~2.6 KFLOP each).
__global__ void head_kernel(const float* __restrict__ h,
                            const float* __restrict__ fc1_w, const float* __restrict__ fc1_b,
                            const float* __restrict__ bn1_g, const float* __restrict__ bn1_b,
                            const float* __restrict__ bn1_m, const float* __restrict__ bn1_v,
                            const float* __restrict__ fc2_w, const float* __restrict__ fc2_b,
                            const float* __restrict__ bn2_g, const float* __restrict__ bn2_b,
                            const float* __restrict__ bn2_m, const float* __restrict__ bn2_v,
                            const float* __restrict__ fc3_w, const float* __restrict__ fc3_b,
                            float* __restrict__ out)
{
    const int b = blockIdx.x * blockDim.x + threadIdx.x;
    if (b >= B_TOT) return;
    const float* hb = h + (size_t)b * H_DIM;

    float t1[32];
    #pragma unroll 4
    for (int o = 0; o < 32; ++o) {
        float s = fc1_b[o];
        for (int k = 0; k < 64; ++k) s += hb[k] * fc1_w[o * 64 + k];
        s = (s - bn1_m[o]) * __builtin_amdgcn_rsqf(bn1_v[o] + 1e-5f) * bn1_g[o] + bn1_b[o];
        t1[o] = s > 0.0f ? s : 0.0f;
    }
    float t2[16];
    #pragma unroll 4
    for (int o = 0; o < 16; ++o) {
        float s = fc2_b[o];
        for (int k = 0; k < 32; ++k) s += t1[k] * fc2_w[o * 32 + k];
        s = (s - bn2_m[o]) * __builtin_amdgcn_rsqf(bn2_v[o] + 1e-5f) * bn2_g[o] + bn2_b[o];
        t2[o] = s > 0.0f ? s : 0.0f;
    }
    float s = fc3_b[0];
    #pragma unroll
    for (int k = 0; k < 16; ++k) s += t2[k] * fc3_w[k];
    out[b] = s;
}

extern "C" void kernel_launch(void* const* d_in, const int* in_sizes, int n_in,
                              void* d_out, int out_size, void* d_ws, size_t ws_size,
                              hipStream_t stream) {
    const float* x     = (const float*)d_in[0];
    const float* Wih0  = (const float*)d_in[1];
    const float* Whh0  = (const float*)d_in[2];
    const float* bih0  = (const float*)d_in[3];
    const float* bhh0  = (const float*)d_in[4];
    const float* Wih1  = (const float*)d_in[5];
    const float* Whh1  = (const float*)d_in[6];
    const float* bih1  = (const float*)d_in[7];
    const float* bhh1  = (const float*)d_in[8];
    const float* fc1_w = (const float*)d_in[9];
    const float* fc1_b = (const float*)d_in[10];
    const float* bn1_g = (const float*)d_in[11];
    const float* bn1_b = (const float*)d_in[12];
    const float* bn1_m = (const float*)d_in[13];
    const float* bn1_v = (const float*)d_in[14];
    const float* fc2_w = (const float*)d_in[15];
    const float* fc2_b = (const float*)d_in[16];
    const float* bn2_g = (const float*)d_in[17];
    const float* bn2_b = (const float*)d_in[18];
    const float* bn2_m = (const float*)d_in[19];
    const float* bn2_v = (const float*)d_in[20];
    const float* fc3_w = (const float*)d_in[21];
    const float* fc3_b = (const float*)d_in[22];

    float* h_last = (float*)d_ws;   // B_TOT * H_DIM f32 = 256 KB

    const size_t shmem = (size_t)3 * G_DIM * H_DIM * sizeof(__bf16)   // bf16 weights
                       + 2 * G_DIM * sizeof(float)                    // biases
                       + G_DIM * IP * sizeof(float)                   // Wih0 padded
                       + BM * IP * sizeof(float)                      // x_t tile
                       + 2 * BM * H_DIM * sizeof(__bf16);             // h1/h2 staging

    lstm2_fused_kernel<<<B_TOT / BM, 128, shmem, stream>>>(
        x, Wih0, Whh0, bih0, bhh0, Wih1, Whh1, bih1, bhh1, h_last);

    head_kernel<<<(B_TOT + 255) / 256, 256, 0, stream>>>(
        h_last, fc1_w, fc1_b, bn1_g, bn1_b, bn1_m, bn1_v,
        fc2_w, fc2_b, bn2_g, bn2_b, bn2_m, bn2_v, fc3_w, fc3_b,
        (float*)d_out);
}
